// DifferentialAttention_9586367004701
// MI455X (gfx1250) — matrix operand order repaired and checked by reading, not compiled, not run on hardware
//
#include <hip/hip_runtime.h>
#include <hip/hip_bf16.h>

typedef __attribute__((ext_vector_type(16))) __bf16 v16bf;
typedef __attribute__((ext_vector_type(8)))  float  v8f;
typedef __attribute__((ext_vector_type(4)))  __bf16 bf16x4;
typedef int v4i __attribute__((vector_size(16)));   // matches builtin param type

#define DIMN     2048
#define QK_SCALE 0.125f                  // 64^-0.5
#define LAM_INIT 0.7836057665316245f     // 0.8 - 0.6*exp(-3.6)
#define RMS_EPS  1e-5f
#define LDT      40                      // padded LDS row stride (bf16 elems)

#if defined(__gfx1250__) && __has_builtin(__builtin_amdgcn_global_load_async_to_lds_b128) && \
    __has_builtin(__builtin_amdgcn_s_wait_asynccnt)
#define USE_ASYNC 1
#else
#define USE_ASYNC 0
#endif

// ---------------- fragment loaders (CDNA5 wave32 WMMA layouts) ----------------
// A 16x32 bf16: lane<16 holds K chunks [0,8)+[16,24); lane>=16 holds [8,16)+[24,32)
__device__ __forceinline__ v16bf ld_frag_a(const __bf16* base, int rowbase, int ld) {
  const int lane = threadIdx.x & 31;
  const __bf16* p = base + (size_t)(rowbase + (lane & 15)) * ld + ((lane & 16) ? 8 : 0);
  union { uint4 u[2]; v16bf v; } c;
  c.u[0] = *reinterpret_cast<const uint4*>(p);
  c.u[1] = *reinterpret_cast<const uint4*>(p + 16);
  return c.v;
}
// B 32x16 bf16 (fed as rows of B^T): lane<16 holds K chunks [0,8)+[16,24); lane>=16 holds [8,16)+[24,32)
__device__ __forceinline__ v16bf ld_frag_b(const __bf16* base, int rowbase, int ld) {
  const int lane = threadIdx.x & 31;
  const __bf16* p = base + (size_t)(rowbase + (lane & 15)) * ld + ((lane & 16) ? 8 : 0);
  union { uint4 u[2]; v16bf v; } c;
  c.u[0] = *reinterpret_cast<const uint4*>(p);
  c.u[1] = *reinterpret_cast<const uint4*>(p + 16);
  return c.v;
}

// copy one 32-elem bf16 row segment (64 B) global -> LDS
__device__ __forceinline__ void copy_row32(const __bf16* g, __bf16* s) {
#if USE_ASYNC
  #pragma unroll
  for (int j = 0; j < 4; j++)
    __builtin_amdgcn_global_load_async_to_lds_b128(
        (__attribute__((address_space(1))) v4i*)(g + j * 8),
        (__attribute__((address_space(3))) v4i*)(s + j * 8), 0, 0);
#else
  const uint4* gu = reinterpret_cast<const uint4*>(g);
  uint4* su = reinterpret_cast<uint4*>(s);
  #pragma unroll
  for (int j = 0; j < 4; j++) su[j] = gu[j];
#endif
}

// ---------------- conversion kernels ----------------
__global__ void cvt_f32_bf16(const float* __restrict__ src, __bf16* __restrict__ dst, int n) {
  int i = blockIdx.x * blockDim.x + threadIdx.x;
  if (i < n) dst[i] = (__bf16)src[i];
}

// W'[r,k] = W[srcrow(r),k]*scale. permute==1: fold the split-qk rearrangement
// (b n (h d q) -> b n (q h) d) into the weight rows so GEMM output is pre-split.
__global__ void cvt_w_perm(const float* __restrict__ src, __bf16* __restrict__ dst,
                           float scale, int permute) {
  int i = blockIdx.x * blockDim.x + threadIdx.x;   // over 2048*2048
  int row = i >> 11, k = i & 2047;
  int srow = row;
  if (permute) {
    int qi = row >> 10, h = (row >> 6) & 15, d = row & 63;
    srow = h * 128 + d * 2 + qi;
  }
  dst[i] = (__bf16)(src[(size_t)srow * DIMN + k] * scale);
}

// ---------------- tiled WMMA GEMM: C[M,N] = A[M,K] * Bw[N,K]^T ----------------
// 128x128 block, 4 waves, each wave 64x64 (16 accum tiles), BK=32, LDS double-buffered,
// async global->LDS prefetch overlapped with WMMA.
template<int F32OUT>
__global__ void __launch_bounds__(128)
gemm_bf16(const __bf16* __restrict__ A, const __bf16* __restrict__ Bw,
          void* __restrict__ Cout, const float* __restrict__ bias,
          int M, int N, int K) {
  __shared__ __bf16 As[2][128 * LDT];
  __shared__ __bf16 Bs[2][128 * LDT];
  const int tid  = threadIdx.x;           // 0..127
  const int lane = tid & 31;
  const int wid  = tid >> 5;              // 0..3
  const int waveM = wid & 1;              // 2 x 64-row slabs
  const int waveN = wid >> 1;             // 2 x 64-col slabs
  const int bm = blockIdx.x * 128;
  const int bn = blockIdx.y * 128;

  v8f acc[4][4];
  #pragma unroll
  for (int mt = 0; mt < 4; mt++)
    #pragma unroll
    for (int nt = 0; nt < 4; nt++)
      #pragma unroll
      for (int e = 0; e < 8; e++) acc[mt][nt][e] = 0.0f;

  const __bf16* garow = A  + (size_t)(bm + tid) * K;  // thread copies one 128-tile row
  const __bf16* gbrow = Bw + (size_t)(bn + tid) * K;

  // prologue: stage tile 0
  copy_row32(garow, &As[0][tid * LDT]);
  copy_row32(gbrow, &Bs[0][tid * LDT]);
#if USE_ASYNC
  __builtin_amdgcn_s_wait_asynccnt(0);
#endif
  __syncthreads();

  for (int k0 = 0; k0 < K; k0 += 32) {
    const int buf = (k0 >> 5) & 1;
    // prefetch next tile into the other buffer (overlaps with WMMA below)
    if (k0 + 32 < K) {
      copy_row32(garow + k0 + 32, &As[buf ^ 1][tid * LDT]);
      copy_row32(gbrow + k0 + 32, &Bs[buf ^ 1][tid * LDT]);
    }

    v16bf bfr[4];
    #pragma unroll
    for (int nt = 0; nt < 4; nt++)
      bfr[nt] = ld_frag_b(&Bs[buf][0], waveN * 64 + nt * 16, LDT);
    #pragma unroll
    for (int mt = 0; mt < 4; mt++) {
      v16bf af = ld_frag_a(&As[buf][0], waveM * 64 + mt * 16, LDT);
      #pragma unroll
      for (int nt = 0; nt < 4; nt++)
        acc[mt][nt] = __builtin_amdgcn_wmma_f32_16x16x32_bf16(
            false, af, false, bfr[nt], (short)0, acc[mt][nt], false, false);
    }

#if USE_ASYNC
    __builtin_amdgcn_s_wait_asynccnt(0);
#endif
    __syncthreads();
  }

  // C layout: VGPR i -> row i (+8 for upper half-wave), col = lane&15
  const int rofs = (lane & 16) >> 1;
  const int cc   = lane & 15;
  #pragma unroll
  for (int mt = 0; mt < 4; mt++)
    #pragma unroll
    for (int nt = 0; nt < 4; nt++)
      #pragma unroll
      for (int i = 0; i < 8; i++) {
        int gr = bm + waveM * 64 + mt * 16 + i + rofs;
        int gc = bn + waveN * 64 + nt * 16 + cc;
        if constexpr (F32OUT) {
          ((float*)Cout)[(size_t)gr * N + gc] = acc[mt][nt][i] + bias[gc];
        } else {
          ((__bf16*)Cout)[(size_t)gr * N + gc] = (__bf16)acc[mt][nt][i];
        }
      }
}

// ---------------- per-token differential attention (1 wave / token) ----------------
__global__ void __launch_bounds__(128)
attn_kernel(const __bf16* __restrict__ qb, const __bf16* __restrict__ kb,
            const __bf16* __restrict__ vb,
            const float* __restrict__ lq1, const float* __restrict__ lq2,
            const float* __restrict__ lk1, const float* __restrict__ lk2,
            __bf16* __restrict__ ab) {
  __shared__ float sdiff[4][256];
  const int lane = threadIdx.x & 31;
  const int w    = threadIdx.x >> 5;
  const int t    = blockIdx.x * 4 + w;

  // lambda = exp(lq1.lk1) - exp(lq2.lk2) + LAM_INIT  (64-dim dots, wave-reduced)
  float s1 = lq1[lane] * lk1[lane] + lq1[lane + 32] * lk1[lane + 32];
  float s2 = lq2[lane] * lk2[lane] + lq2[lane + 32] * lk2[lane + 32];
  for (int m = 1; m < 32; m <<= 1) { s1 += __shfl_xor(s1, m, 32); s2 += __shfl_xor(s2, m, 32); }
  const float lam = __expf(s1) - __expf(s2) + LAM_INIT;

  // att(32x32) = Q(32x64) * K^T : 2x2 tiles, 2 K-chunks of 32
  const __bf16* qt = qb + (size_t)t * DIMN;   // [32 rows][64], scale pre-folded
  const __bf16* kt = kb + (size_t)t * DIMN;
  v8f att[2][2];
  #pragma unroll
  for (int mt = 0; mt < 2; mt++)
    #pragma unroll
    for (int nt = 0; nt < 2; nt++)
      #pragma unroll
      for (int e = 0; e < 8; e++) att[mt][nt][e] = 0.0f;
  #pragma unroll
  for (int kc = 0; kc < 2; kc++) {
    v16bf aq[2], bk[2];
    #pragma unroll
    for (int mt = 0; mt < 2; mt++) aq[mt] = ld_frag_a(qt + kc * 32, mt * 16, 64);
    #pragma unroll
    for (int nt = 0; nt < 2; nt++) bk[nt] = ld_frag_b(kt + kc * 32, nt * 16, 64);
    #pragma unroll
    for (int mt = 0; mt < 2; mt++)
      #pragma unroll
      for (int nt = 0; nt < 2; nt++)
        att[mt][nt] = __builtin_amdgcn_wmma_f32_16x16x32_bf16(
            false, aq[mt], false, bk[nt], (short)0, att[mt][nt], false, false);
  }
  // nan_to_num
  #pragma unroll
  for (int mt = 0; mt < 2; mt++)
    #pragma unroll
    for (int nt = 0; nt < 2; nt++)
      #pragma unroll
      for (int e = 0; e < 8; e++) {
        float v = att[mt][nt][e];
        att[mt][nt][e] = (v == v) ? v : 0.0f;
      }

  // row softmax over 32 cols (cols live across a 16-lane group), keep diagonal blocks,
  // diff = att[:16,:16] - lam * att[16:,16:]
  float d00[8], diffv[8];
  #pragma unroll
  for (int mt = 0; mt < 2; mt++) {
    float mx[8], e0[8], e1[8], s[8];
    #pragma unroll
    for (int i = 0; i < 8; i++) mx[i] = fmaxf(att[mt][0][i], att[mt][1][i]);
    #pragma unroll
    for (int i = 0; i < 8; i++)
      for (int m = 1; m < 16; m <<= 1) mx[i] = fmaxf(mx[i], __shfl_xor(mx[i], m, 32));
    #pragma unroll
    for (int i = 0; i < 8; i++) {
      e0[i] = __expf(att[mt][0][i] - mx[i]);
      e1[i] = __expf(att[mt][1][i] - mx[i]);
      s[i]  = e0[i] + e1[i];
    }
    #pragma unroll
    for (int i = 0; i < 8; i++)
      for (int m = 1; m < 16; m <<= 1) s[i] += __shfl_xor(s[i], m, 32);
    #pragma unroll
    for (int i = 0; i < 8; i++) {
      float nv = (mt == 0 ? e0[i] : e1[i]) / s[i];
      if (mt == 0) d00[i] = nv; else diffv[i] = d00[i] - lam * nv;
    }
  }

  // stage diff(16x16) to LDS (C layout -> [a][h])
  const int rofs = (lane & 16) >> 1;
  const int hcol = lane & 15;
  #pragma unroll
  for (int i = 0; i < 8; i++) sdiff[w][(i + rofs) * 16 + hcol] = diffv[i];
  __syncthreads();

  // out[a, d2] = sum_h diff[a,h] * v[h,d2] ; lane owns d2 = 4*lane..4*lane+3
  const __bf16* vt = vb + (size_t)t * DIMN;   // [16 h][128 d2]
  float out4[16][4];
  #pragma unroll
  for (int a = 0; a < 16; a++)
    #pragma unroll
    for (int c = 0; c < 4; c++) out4[a][c] = 0.0f;
  for (int h = 0; h < 16; h++) {
    bf16x4 v4 = *reinterpret_cast<const bf16x4*>(vt + h * 128 + lane * 4);
    float v0 = (float)v4[0], v1 = (float)v4[1], v2 = (float)v4[2], v3 = (float)v4[3];
    #pragma unroll
    for (int a = 0; a < 16; a++) {
      float wgt = sdiff[w][a * 16 + h];   // uniform address -> LDS broadcast
      out4[a][0] += wgt * v0; out4[a][1] += wgt * v1;
      out4[a][2] += wgt * v2; out4[a][3] += wgt * v3;
    }
  }

  // RMSNorm over d2 (128) per row a, * (1 - LAM_INIT), store bf16
  #pragma unroll
  for (int a = 0; a < 16; a++) {
    float ss = out4[a][0]*out4[a][0] + out4[a][1]*out4[a][1]
             + out4[a][2]*out4[a][2] + out4[a][3]*out4[a][3];
    for (int m = 1; m < 32; m <<= 1) ss += __shfl_xor(ss, m, 32);
    float sc = rsqrtf(ss * (1.0f / 128.0f) + RMS_EPS) * (1.0f - LAM_INIT);
    bf16x4 o;
    o[0] = (__bf16)(out4[a][0] * sc); o[1] = (__bf16)(out4[a][1] * sc);
    o[2] = (__bf16)(out4[a][2] * sc); o[3] = (__bf16)(out4[a][3] * sc);
    *reinterpret_cast<bf16x4*>(ab + (size_t)t * DIMN + a * 128 + lane * 4) = o;
  }
}

// ---------------- launch ----------------
extern "C" void kernel_launch(void* const* d_in, const int* in_sizes, int n_in,
                              void* d_out, int out_size, void* d_ws, size_t ws_size,
                              hipStream_t stream) {
  const float* x   = (const float*)d_in[0];
  const float* Wq  = (const float*)d_in[1];
  const float* Wk  = (const float*)d_in[2];
  const float* Wv  = (const float*)d_in[3];
  const float* Wo  = (const float*)d_in[4];
  const float* bo  = (const float*)d_in[5];
  const float* lq1 = (const float*)d_in[6];
  const float* lq2 = (const float*)d_in[7];
  const float* lk1 = (const float*)d_in[8];
  const float* lk2 = (const float*)d_in[9];
  const int M = in_sizes[0] / DIMN;     // 16384 tokens

  // workspace layout (bf16): xb | wq wk wv wo | qb kb vb ; attn-out aliases xb
  char* ws = (char*)d_ws;
  const size_t xbytes = (size_t)M * DIMN * sizeof(__bf16);   // 67 MB
  const size_t wbytes = (size_t)DIMN * DIMN * sizeof(__bf16);
  __bf16* xb  = (__bf16*)ws;
  __bf16* wqb = (__bf16*)(ws + xbytes);
  __bf16* wkb = wqb + (size_t)DIMN * DIMN;
  __bf16* wvb = wkb + (size_t)DIMN * DIMN;
  __bf16* wob = wvb + (size_t)DIMN * DIMN;
  __bf16* qb  = (__bf16*)(ws + xbytes + 4 * wbytes);
  __bf16* kb  = qb + (size_t)M * DIMN;
  __bf16* vb  = kb + (size_t)M * DIMN;
  __bf16* ab  = xb;   // xb fully consumed by the 3 QKV GEMMs before attn writes it

  const int nx = M * DIMN;
  cvt_f32_bf16<<<nx / 256, 256, 0, stream>>>(x, xb, nx);
  const int nw = DIMN * DIMN;
  cvt_w_perm<<<nw / 256, 256, 0, stream>>>(Wq, wqb, QK_SCALE, 1);  // fold scale+split
  cvt_w_perm<<<nw / 256, 256, 0, stream>>>(Wk, wkb, 1.0f, 1);      // fold split
  cvt_w_perm<<<nw / 256, 256, 0, stream>>>(Wv, wvb, 1.0f, 0);
  cvt_w_perm<<<nw / 256, 256, 0, stream>>>(Wo, wob, 1.0f, 0);

  dim3 g(M / 128, DIMN / 128);
  gemm_bf16<0><<<g, 128, 0, stream>>>(xb, wqb, qb, nullptr, M, DIMN, DIMN);
  gemm_bf16<0><<<g, 128, 0, stream>>>(xb, wkb, kb, nullptr, M, DIMN, DIMN);
  gemm_bf16<0><<<g, 128, 0, stream>>>(xb, wvb, vb, nullptr, M, DIMN, DIMN);

  attn_kernel<<<M / 4, 128, 0, stream>>>(qb, kb, vb, lq1, lq2, lk1, lk2, ab);

  gemm_bf16<1><<<g, 128, 0, stream>>>(ab, wob, d_out, bo, M, DIMN, DIMN);
}
